// Attention_40475771797639
// MI455X (gfx1250) — matrix operand order repaired and checked by reading, not compiled, not run on hardware
//
#include <hip/hip_runtime.h>

typedef __attribute__((ext_vector_type(16))) _Float16 v16h;
typedef __attribute__((ext_vector_type(8)))  _Float16 v8h;
typedef __attribute__((ext_vector_type(8)))  float    v8f;
typedef __attribute__((ext_vector_type(4)))  unsigned v4u;
typedef __attribute__((ext_vector_type(8)))  int      v8i;
typedef __attribute__((ext_vector_type(4)))  int      v4i;

static constexpr int kDIM  = 1024;
static constexpr int kNH   = 16;
static constexpr int kDH   = 64;
static constexpr int kB    = 4;
static constexpr int kS    = 2048;
static constexpr float kNegScale = -0.125f;    // -1/sqrt(64)

__device__ __forceinline__ v8f wmma_f16(v16h a, v16h b, v8f c) {
  return __builtin_amdgcn_wmma_f32_16x16x32_f16(
      false, a, false, b, (short)0, c, false, false);
}

// ---------------- TDM: tensor_load_to_lds with hand-packed D# ----------------
// 2D tile of 2-byte elements: tile_d0 x tile_d1, row stride stride0 (elements).
__device__ __forceinline__ void tdm_load_2d(unsigned ldsAddr, const void* gptr,
                                            unsigned tensor_d0, unsigned tensor_d1,
                                            unsigned tile_d0, unsigned tile_d1,
                                            unsigned long long stride0) {
  unsigned long long ga = (unsigned long long)(size_t)gptr;
  v4u g0;
  g0[0] = 1u;                                   // count=1, user load descriptor
  g0[1] = ldsAddr;                              // lds_addr [63:32]
  g0[2] = (unsigned)ga;                         // global_addr[31:0]
  g0[3] = ((unsigned)(ga >> 32) & 0x01FFFFFFu)  // global_addr[56:32]
          | (2u << 30);                         // type=2 ("image")
  v8i g1;
  g1[0] = (int)(1u << 16);                      // data_size=1 -> 2 bytes
  g1[1] = (int)((tensor_d0 & 0xFFFFu) << 16);   // [63:48] tensor_dim0 lo
  g1[2] = (int)((tensor_d0 >> 16) | ((tensor_d1 & 0xFFFFu) << 16));
  g1[3] = (int)((tensor_d1 >> 16) | (tile_d0 << 16));   // [127:112] tile_dim0
  g1[4] = (int)(tile_d1 & 0xFFFFu);             // tile_dim1 (tile_dim2=0)
  g1[5] = (int)(unsigned)(stride0 & 0xFFFFFFFFu);       // tensor_dim0_stride lo
  g1[6] = (int)((unsigned)(stride0 >> 32) & 0xFFFFu);   // stride hi; dim1_stride=0
  g1[7] = 0;
  v4i z4 = {0, 0, 0, 0};
#if defined(__clang_major__) && __clang_major__ >= 23
  v8i z8 = {0, 0, 0, 0, 0, 0, 0, 0};
  __builtin_amdgcn_tensor_load_to_lds(g0, g1, z4, z4, z8, 0);
#else
  __builtin_amdgcn_tensor_load_to_lds(g0, g1, z4, z4, 0);
#endif
}

// ---------------- fp32 -> f16 weight conversion ----------------
__global__ void cvt_f32_to_f16(const float* __restrict__ src,
                               _Float16* __restrict__ dst, int n) {
  int i = blockIdx.x * blockDim.x + threadIdx.x;
  if (i < n) dst[i] = (_Float16)src[i];
}

// ---------------- operand load helpers ----------------
__device__ __forceinline__ v16h load_a_from_f32(const float* p, int ac0, int ac1) {
  const float4* p0 = (const float4*)(p + ac0);
  const float4* p1 = (const float4*)(p + ac1);
  float4 u0 = p0[0], u1 = p0[1], w0 = p1[0], w1 = p1[1];
  v16h a;
  a[0]=(_Float16)u0.x;  a[1]=(_Float16)u0.y;  a[2]=(_Float16)u0.z;  a[3]=(_Float16)u0.w;
  a[4]=(_Float16)u1.x;  a[5]=(_Float16)u1.y;  a[6]=(_Float16)u1.z;  a[7]=(_Float16)u1.w;
  a[8]=(_Float16)w0.x;  a[9]=(_Float16)w0.y;  a[10]=(_Float16)w0.z; a[11]=(_Float16)w0.w;
  a[12]=(_Float16)w1.x; a[13]=(_Float16)w1.y; a[14]=(_Float16)w1.z; a[15]=(_Float16)w1.w;
  return a;
}
__device__ __forceinline__ v16h load_a_f16(const _Float16* p, int ac0, int ac1) {
  v8h lo = *(const v8h*)(p + ac0);
  v8h hi = *(const v8h*)(p + ac1);
  return __builtin_shufflevector(lo, hi, 0,1,2,3,4,5,6,7,8,9,10,11,12,13,14,15);
}
__device__ __forceinline__ void load_b4(const _Float16* wbase, int n0, int ln, v16h b[4]) {
#pragma unroll
  for (int j = 0; j < 4; ++j)
    b[j] = *(const v16h*)(wbase + (size_t)(n0 + j * 16 + ln) * kDIM);
}

// ---------------- projection GEMM: Y = scale*(X @ W^T + b) ----------------
// Depth-2 ping-pong pipeline: two operand buffer sets, no register rotation.
__global__ __launch_bounds__(256) void proj_gemm(
    const float* __restrict__ X, const _Float16* __restrict__ W16,
    const float* __restrict__ bias, _Float16* __restrict__ Y,
    float out_scale, int mode) {
  const int lane = threadIdx.x & 31;
  const int wave = threadIdx.x >> 5;
  const int gw   = blockIdx.x * 8 + wave;
  const int mt   = gw >> 4, nt = gw & 15;
  const int m0 = mt * 16, n0 = nt * 64;
  const int g = lane >> 4, ln = lane & 15;

  const float* arow = X + (size_t)(m0 + ln) * kDIM;
  const int ac0 = g * 16, ac1 = g * 16 + 8;

  v8f acc[4] = {};
  v16h a0 = load_a_from_f32(arow, ac0, ac1);
  v16h b0[4]; load_b4(W16 + g * 16, n0, ln, b0);
  v16h a1 = load_a_from_f32(arow + 32, ac0, ac1);
  v16h b1[4]; load_b4(W16 + 32 + g * 16, n0, ln, b1);
  for (int k = 64; k < kDIM; k += 64) {
    __builtin_prefetch(arow + k + 64 + ac0, 0, 3);
#pragma unroll
    for (int j = 0; j < 4; ++j) acc[j] = wmma_f16(a0, b0[j], acc[j]);
    a0 = load_a_from_f32(arow + k, ac0, ac1);
    load_b4(W16 + k + g * 16, n0, ln, b0);
#pragma unroll
    for (int j = 0; j < 4; ++j) acc[j] = wmma_f16(a1, b1[j], acc[j]);
    a1 = load_a_from_f32(arow + k + 32, ac0, ac1);
    load_b4(W16 + k + 32 + g * 16, n0, ln, b1);
  }
#pragma unroll
  for (int j = 0; j < 4; ++j) acc[j] = wmma_f16(a0, b0[j], acc[j]);
#pragma unroll
  for (int j = 0; j < 4; ++j) acc[j] = wmma_f16(a1, b1[j], acc[j]);

  // epilogue: bias + scale + head-layout scatter (f16)
#pragma unroll
  for (int j = 0; j < 4; ++j) {
    const int col = n0 + j * 16 + ln;
    const float bv = bias[col];
    const int hh = col >> 6, dd = col & 63;
#pragma unroll
    for (int r = 0; r < 8; ++r) {
      const int row = m0 + r + 8 * g;
      const int bb = row >> 11, ss = row & 2047;
      const _Float16 hv = (_Float16)(out_scale * (acc[j][r] + bv));
      size_t idx;
      if (mode == 0) idx = (((size_t)(bb * kNH + hh)) * kS + ss) * kDH + dd;
      else           idx = (((size_t)(bb * kNH + hh)) * kDH + dd) * (size_t)kS + ss;
      Y[idx] = hv;
    }
  }
}

// ---------------- flash attention with TDM-staged K/V ----------------
// Block = one (b,h) head, 8 waves = 8 q-tiles sharing LDS K/V tiles.
// Q pre-scaled by -1/sqrt(Dh). Q,K: (BH,S,Dh) f16. VT: (BH,Dh,S) f16.
__global__ __launch_bounds__(256) void flash_attn(
    const _Float16* __restrict__ Q, const _Float16* __restrict__ K,
    const _Float16* __restrict__ VT, _Float16* __restrict__ AO) {
  __shared__ __align__(32) _Float16 sK[2][32 * 64];  // kv-local row major
  __shared__ __align__(32) _Float16 sV[2][64 * 32];  // d row major
  __shared__ __align__(32) _Float16 sP[8][16 * 32];  // per-wave P staging
  const int lane = threadIdx.x & 31;
  const int wave = threadIdx.x >> 5;
  const int bh   = blockIdx.x >> 4;          // 64 heads
  const int q0   = (blockIdx.x & 15) * 128 + wave * 16;
  const int g = lane >> 4, ln = lane & 15;

  const _Float16* kbase = K + (size_t)bh * kS * kDH;
  const _Float16* vbase = VT + (size_t)bh * kDH * kS;

  // Q as two 16x32 A-operands (d 0..31 and 32..63)
  const _Float16* qbase = Q + ((size_t)bh * kS + q0 + ln) * kDH;
  v16h qa0 = load_a_f16(qbase, g * 16, g * 16 + 8);
  v16h qa1 = load_a_f16(qbase + 32, g * 16, g * 16 + 8);

  float mrow[8], lrow[8];
  v8f o[4] = {};
#pragma unroll
  for (int r = 0; r < 8; ++r) { mrow[r] = -__builtin_huge_valf(); lrow[r] = 0.f; }

  _Float16* sp = &sP[wave][0];
  const int nTiles = kS / 32;

  // preload tile 0 (wave 0 drives the Tensor Data Mover)
  if (wave == 0) {
    tdm_load_2d((unsigned)(size_t)&sK[0][0], kbase, 2048, 1, 2048, 1, 2048ull);
    tdm_load_2d((unsigned)(size_t)&sV[0][0], vbase, kS, kDH, 32, kDH, (unsigned long long)kS);
  }

  for (int t = 0; t < nTiles; ++t) {
    const int kv0 = t * 32;
    const int buf = t & 1;
    if (wave == 0) {
      if (t + 1 < nTiles) {
        tdm_load_2d((unsigned)(size_t)&sK[buf ^ 1][0],
                    kbase + (size_t)(kv0 + 32) * kDH, 2048, 1, 2048, 1, 2048ull);
        tdm_load_2d((unsigned)(size_t)&sV[buf ^ 1][0],
                    vbase + (kv0 + 32), kS, kDH, 32, kDH, (unsigned long long)kS);
        __builtin_amdgcn_s_wait_tensorcnt(2);   // tile t landed, t+1 in flight
      } else {
        __builtin_amdgcn_s_wait_tensorcnt(0);
      }
    }
    __syncthreads();                            // tile t visible to all waves

    // scores: two 16x16 tiles from LDS-staged K
    const _Float16* kt = &sK[buf][0];
    v16h kb[2][2];
#pragma unroll
    for (int h2 = 0; h2 < 2; ++h2) {
      const _Float16* kr = kt + (h2 * 16 + ln) * kDH + g * 16;
      kb[h2][0] = *(const v16h*)(kr);
      kb[h2][1] = *(const v16h*)(kr + 32);
    }
    v8f s0 = {}, s1 = {};
    s0 = wmma_f16(qa0, kb[0][0], s0); s0 = wmma_f16(qa1, kb[0][1], s0);
    s1 = wmma_f16(qa0, kb[1][0], s1); s1 = wmma_f16(qa1, kb[1][1], s1);

    // online softmax (row r lives at vgpr r, lanes (row/8)*16..+15)
#pragma unroll
    for (int r = 0; r < 8; ++r) {
      float tmax = fmaxf(s0[r], s1[r]);
      tmax = fmaxf(tmax, __shfl_xor(tmax, 1, 32));
      tmax = fmaxf(tmax, __shfl_xor(tmax, 2, 32));
      tmax = fmaxf(tmax, __shfl_xor(tmax, 4, 32));
      tmax = fmaxf(tmax, __shfl_xor(tmax, 8, 32));
      const float mn = fmaxf(mrow[r], tmax);
      const float alpha = __expf(mrow[r] - mn);
      mrow[r] = mn;
      const float p0 = __expf(s0[r] - mn);
      const float p1 = __expf(s1[r] - mn);
      s0[r] = p0; s1[r] = p1;
      float rs = p0 + p1;
      rs += __shfl_xor(rs, 1, 32);
      rs += __shfl_xor(rs, 2, 32);
      rs += __shfl_xor(rs, 4, 32);
      rs += __shfl_xor(rs, 8, 32);
      lrow[r] = lrow[r] * alpha + rs;
#pragma unroll
      for (int j = 0; j < 4; ++j) o[j][r] *= alpha;
    }

    // P: C-layout f32 -> A-layout f16 via per-wave LDS slab (same-wave in-order)
#pragma unroll
    for (int r = 0; r < 8; ++r) {
      const int row = r + 8 * g;
      sp[row * 32 + ln]      = (_Float16)s0[r];
      sp[row * 32 + 16 + ln] = (_Float16)s1[r];
    }
    v8h plo = *(const v8h*)(sp + ln * 32 + g * 16);
    v8h phi = *(const v8h*)(sp + ln * 32 + g * 16 + 8);
    v16h pa = __builtin_shufflevector(plo, phi, 0,1,2,3,4,5,6,7,8,9,10,11,12,13,14,15);

    // O += P @ V from LDS-staged V^T
    const _Float16* vt = &sV[buf][0];
#pragma unroll
    for (int j = 0; j < 4; ++j) {
      v16h vb = *(const v16h*)(vt + (j * 16 + ln) * 32 + g * 16);
      o[j] = wmma_f16(pa, vb, o[j]);
    }
    __syncthreads();                            // all waves done with buffer t
  }

  // normalize and merge heads back into (B,S,DIM)
  const int b = bh >> 4, h = bh & 15;
#pragma unroll
  for (int r = 0; r < 8; ++r) {
    const float inv = 1.0f / lrow[r];
    const int srow = q0 + r + 8 * g;
    const size_t mr = (size_t)(b * kS + srow);
#pragma unroll
    for (int j = 0; j < 4; ++j) {
      const int col = h * kDH + j * 16 + ln;
      AO[mr * kDIM + col] = (_Float16)(o[j][r] * inv);
    }
  }
}

// ---------------- output projection: out = AO @ Wp^T + bp (f32 out) --------
__global__ __launch_bounds__(256) void out_gemm(
    const _Float16* __restrict__ A, const _Float16* __restrict__ W16,
    const float* __restrict__ bias, float* __restrict__ Y) {
  const int lane = threadIdx.x & 31;
  const int wave = threadIdx.x >> 5;
  const int gw = blockIdx.x * 8 + wave;
  const int mt = gw >> 4, nt = gw & 15;
  const int m0 = mt * 16, n0 = nt * 64;
  const int g = lane >> 4, ln = lane & 15;

  const _Float16* arow = A + (size_t)(m0 + ln) * kDIM;
  const int ac0 = g * 16, ac1 = g * 16 + 8;

  v8f acc[4] = {};
  v16h a0 = load_a_f16(arow, ac0, ac1);
  v16h b0[4]; load_b4(W16 + g * 16, n0, ln, b0);
  v16h a1 = load_a_f16(arow + 32, ac0, ac1);
  v16h b1[4]; load_b4(W16 + 32 + g * 16, n0, ln, b1);
  for (int k = 64; k < kDIM; k += 64) {
    __builtin_prefetch(arow + k + 64 + ac0, 0, 3);
#pragma unroll
    for (int j = 0; j < 4; ++j) acc[j] = wmma_f16(a0, b0[j], acc[j]);
    a0 = load_a_f16(arow + k, ac0, ac1);
    load_b4(W16 + k + g * 16, n0, ln, b0);
#pragma unroll
    for (int j = 0; j < 4; ++j) acc[j] = wmma_f16(a1, b1[j], acc[j]);
    a1 = load_a_f16(arow + k + 32, ac0, ac1);
    load_b4(W16 + k + 32 + g * 16, n0, ln, b1);
  }
#pragma unroll
  for (int j = 0; j < 4; ++j) acc[j] = wmma_f16(a0, b0[j], acc[j]);
#pragma unroll
  for (int j = 0; j < 4; ++j) acc[j] = wmma_f16(a1, b1[j], acc[j]);

#pragma unroll
  for (int j = 0; j < 4; ++j) {
    const int col = n0 + j * 16 + ln;
    const float bv = bias[col];
#pragma unroll
    for (int r = 0; r < 8; ++r) {
      const int row = m0 + r + 8 * g;
      Y[(size_t)row * kDIM + col] = acc[j][r] + bv;
    }
  }
}

extern "C" void kernel_launch(void* const* d_in, const int* in_sizes, int n_in,
                              void* d_out, int out_size, void* d_ws, size_t ws_size,
                              hipStream_t stream) {
  (void)in_sizes; (void)n_in; (void)out_size; (void)ws_size;
  const float* x0 = (const float*)d_in[0];
  const float* x1 = (const float*)d_in[1];
  const float* x2 = (const float*)d_in[2];
  const float* Wq = (const float*)d_in[3];
  const float* bq = (const float*)d_in[4];
  const float* Wk = (const float*)d_in[5];
  const float* bk = (const float*)d_in[6];
  const float* Wv = (const float*)d_in[7];
  const float* bv = (const float*)d_in[8];
  const float* Wp = (const float*)d_in[9];
  const float* bp = (const float*)d_in[10];
  float* out = (float*)d_out;

  char* ws = (char*)d_ws;
  const size_t MB = 1024ull * 1024ull;
  _Float16* q16  = (_Float16*)(ws);             // 16 MB (B,H,S,Dh)
  _Float16* k16  = (_Float16*)(ws + 16 * MB);   // 16 MB (B,H,S,Dh)
  _Float16* vt16 = (_Float16*)(ws + 32 * MB);   // 16 MB (B,H,Dh,S)
  _Float16* ao16 = (_Float16*)(ws + 48 * MB);   // 16 MB (B,S,DIM)
  _Float16* wq16 = (_Float16*)(ws + 64 * MB);   // 2 MB each
  _Float16* wk16 = (_Float16*)(ws + 66 * MB);
  _Float16* wv16 = (_Float16*)(ws + 68 * MB);
  _Float16* wp16 = (_Float16*)(ws + 70 * MB);

  const int nW = kDIM * kDIM;
  const int cvtBlocks = (nW + 255) / 256;
  cvt_f32_to_f16<<<cvtBlocks, 256, 0, stream>>>(Wq, wq16, nW);
  cvt_f32_to_f16<<<cvtBlocks, 256, 0, stream>>>(Wk, wk16, nW);
  cvt_f32_to_f16<<<cvtBlocks, 256, 0, stream>>>(Wv, wv16, nW);
  cvt_f32_to_f16<<<cvtBlocks, 256, 0, stream>>>(Wp, wp16, nW);

  proj_gemm<<<1024, 256, 0, stream>>>(x0, wq16, bq, q16, kNegScale, 0);
  proj_gemm<<<1024, 256, 0, stream>>>(x1, wk16, bk, k16, 1.0f, 0);
  proj_gemm<<<1024, 256, 0, stream>>>(x2, wv16, bv, vt16, 1.0f, 1);

  flash_attn<<<1024, 256, 0, stream>>>(q16, k16, vt16, ao16);

  out_gemm<<<1024, 256, 0, stream>>>(ao16, wp16, bp, out);
}
